// MultiHeadedAttention_82944408420463
// MI455X (gfx1250) — matrix operand order repaired and checked by reading, not compiled, not run on hardware
//
#include <hip/hip_runtime.h>
#include <hip/hip_bf16.h>
#include <math.h>

#define B_  8
#define S_  1024
#define E_  1024
#define H_  16
#define HD_ 64
#define M_  (B_*S_)

typedef __attribute__((ext_vector_type(16))) __bf16        v16bf;
typedef __attribute__((ext_vector_type(2)))  __bf16        v2bf;
typedef __attribute__((ext_vector_type(8)))  float         v8f;
typedef __attribute__((ext_vector_type(4)))  unsigned int  u32x4;
typedef __attribute__((ext_vector_type(4)))  float         f32x4;

union Frag16 {
  v16bf          v;
  u32x4          q[2];
  unsigned int   w[8];
  unsigned short u[16];
};

__device__ __forceinline__ unsigned short f32_bf16(float f) {
  unsigned int x = __float_as_uint(f);
  x += 0x7FFFu + ((x >> 16) & 1u);           // round-to-nearest-even
  return (unsigned short)(x >> 16);
}

// pack two fp32 into packed bf16x2 (hardware cvt if available)
__device__ __forceinline__ unsigned int pk2(float x, float y) {
#if __has_builtin(__builtin_amdgcn_cvt_pk_bf16_f32)
  union { v2bf v; unsigned int u; } cv;
  cv.v = __builtin_amdgcn_cvt_pk_bf16_f32(x, y);
  return cv.u;
#else
  return (unsigned int)f32_bf16(x) | ((unsigned int)f32_bf16(y) << 16);
#endif
}

__device__ __forceinline__ void pack16(Frag16& fr, f32x4 f0, f32x4 f1, f32x4 f2, f32x4 f3) {
  fr.w[0] = pk2(f0.x, f0.y);  fr.w[1] = pk2(f0.z, f0.w);
  fr.w[2] = pk2(f1.x, f1.y);  fr.w[3] = pk2(f1.z, f1.w);
  fr.w[4] = pk2(f2.x, f2.y);  fr.w[5] = pk2(f2.z, f2.w);
  fr.w[6] = pk2(f3.x, f3.y);  fr.w[7] = pk2(f3.z, f3.w);
}

// ---------------------------------------------------------------- fp32 -> bf16
__global__ void cvt_f32_to_bf16_kernel(const float* __restrict__ src,
                                       unsigned int* __restrict__ dst, int npairs) {
  int i = blockIdx.x * blockDim.x + threadIdx.x;
  if (i < npairs) {
    f32x4* unused; (void)unused;
    float a = src[2 * i], b = src[2 * i + 1];
    dst[i] = pk2(a, b);
  }
}

// ------------------------------------------------- generic GEMM: C = A*B^T + b
// A: M x K (fp32 or bf16, row-major). Bw: N x K bf16 row-major. C: M x N.
// Each wave computes a 64x64 tile (4x4 accumulators of 16x16).
template<bool A_IS_F32, bool OUT_F32>
__global__ __launch_bounds__(128)
void gemm_bf16_kernel(const void* __restrict__ Aptr,
                      const unsigned short* __restrict__ Bw,
                      const float* __restrict__ bias,
                      void* __restrict__ Cptr,
                      int Kdim, int Ndim) {
  const int lane = threadIdx.x & 31;
  const int wave = threadIdx.x >> 5;
  const int lo = lane & 15, hi = lane >> 4;
  const int tileM = blockIdx.x * 128 + (wave >> 1) * 64;
  const int tileN = blockIdx.y * 128 + (wave & 1) * 64;

  v8f acc[4][4] = {};
  for (int k0 = 0; k0 < Kdim; k0 += 32) {
    Frag16 af[4], bf[4];
#pragma unroll
    for (int i = 0; i < 4; ++i) {
      const int m = tileM + i * 16 + lo;
      if (A_IS_F32) {
        const float* row = (const float*)Aptr + (size_t)m * Kdim + k0 + hi * 8;
        f32x4 f0 = ((const f32x4*)row)[0];
        f32x4 f1 = ((const f32x4*)row)[1];
        f32x4 f2 = ((const f32x4*)(row + 16))[0];
        f32x4 f3 = ((const f32x4*)(row + 16))[1];
        pack16(af[i], f0, f1, f2, f3);
      } else {
        const unsigned short* row =
            (const unsigned short*)Aptr + (size_t)m * Kdim + k0 + hi * 8;
        af[i].q[0] = *(const u32x4*)row;          // K elems 0..7 of fragment
        af[i].q[1] = *(const u32x4*)(row + 16);   // K elems 8..15
      }
    }
#pragma unroll
    for (int j = 0; j < 4; ++j) {
      const int n = tileN + j * 16 + lo;
      const unsigned short* row = Bw + (size_t)n * Kdim + k0 + hi * 8;
      bf[j].q[0] = *(const u32x4*)row;            // K {8h..8h+7}
      bf[j].q[1] = *(const u32x4*)(row + 16);     // K {16+8h..16+8h+7}
    }
#pragma unroll
    for (int i = 0; i < 4; ++i)
#pragma unroll
      for (int j = 0; j < 4; ++j)
        acc[i][j] = __builtin_amdgcn_wmma_f32_16x16x32_bf16(
            false, af[i].v, false, bf[j].v, (short)0, acc[i][j], false, false);
  }
#pragma unroll
  for (int i = 0; i < 4; ++i)
#pragma unroll
    for (int j = 0; j < 4; ++j) {
      const int n = tileN + j * 16 + lo;
      const float bv = bias[n];
#pragma unroll
      for (int r = 0; r < 8; ++r) {
        const int m = tileM + i * 16 + hi * 8 + r;
        const float val = acc[i][j][r] + bv;
        if (OUT_F32) ((float*)Cptr)[(size_t)m * Ndim + n] = val;
        else ((unsigned short*)Cptr)[(size_t)m * Ndim + n] = f32_bf16(val);
      }
    }
}

// -------------------------------------------- scores = (Q K^T)/sqrt(d), masked
__global__ __launch_bounds__(128)
void scores_kernel(const unsigned short* __restrict__ Qb,
                   const unsigned short* __restrict__ Kb,
                   const float* __restrict__ mask,
                   float* __restrict__ attn) {
  const int bh = blockIdx.z, bb = bh / H_, h = bh % H_;
  const int lane = threadIdx.x & 31, wave = threadIdx.x >> 5;
  const int lo = lane & 15, hi = lane >> 4;
  const int tileQ = blockIdx.x * 128 + (wave >> 1) * 64;
  const int tileK = blockIdx.y * 128 + (wave & 1) * 64;
  const unsigned short* Qbase = Qb + (size_t)bb * S_ * E_ + h * HD_;
  const unsigned short* Kbase = Kb + (size_t)bb * S_ * E_ + h * HD_;

  v8f acc[4][4] = {};
#pragma unroll
  for (int k0 = 0; k0 < HD_; k0 += 32) {
    Frag16 af[4], bf[4];
#pragma unroll
    for (int i = 0; i < 4; ++i) {
      const unsigned short* row = Qbase + (size_t)(tileQ + i * 16 + lo) * E_ + k0 + hi * 8;
      af[i].q[0] = *(const u32x4*)row;
      af[i].q[1] = *(const u32x4*)(row + 16);
    }
#pragma unroll
    for (int j = 0; j < 4; ++j) {
      const unsigned short* row = Kbase + (size_t)(tileK + j * 16 + lo) * E_ + k0 + hi * 8;
      bf[j].q[0] = *(const u32x4*)row;
      bf[j].q[1] = *(const u32x4*)(row + 16);
    }
#pragma unroll
    for (int i = 0; i < 4; ++i)
#pragma unroll
      for (int j = 0; j < 4; ++j)
        acc[i][j] = __builtin_amdgcn_wmma_f32_16x16x32_bf16(
            false, af[i].v, false, bf[j].v, (short)0, acc[i][j], false, false);
  }
  const float inv = 0.125f;  // 1/sqrt(64)
#pragma unroll
  for (int i = 0; i < 4; ++i)
#pragma unroll
    for (int j = 0; j < 4; ++j) {
      const int kc = tileK + j * 16 + lo;
#pragma unroll
      for (int r = 0; r < 8; ++r) {
        const int qr = tileQ + i * 16 + hi * 8 + r;
        float s = acc[i][j][r] * inv;
        const float mv = mask[((size_t)bb * S_ + qr) * S_ + kc];
        s = (mv == 0.0f) ? -INFINITY : s * mv;
        attn[((size_t)bh * S_ + qr) * S_ + kc] = s;
      }
    }
}

// ----------------------------- row softmax + dist_score + gnn_adj, in place
__global__ __launch_bounds__(256)
void softmax_add_kernel(float* __restrict__ attn,
                        const float* __restrict__ dist,
                        const float* __restrict__ gnn) {
  const int row = blockIdx.x;            // (b*H + h)*S + q
  const int b = row / (H_ * S_);
  const int q = row % S_;
  float* rp = attn + (size_t)row * S_;
  const float* dp = dist + ((size_t)b * S_ + q) * S_;
  const float* gp = gnn + ((size_t)b * S_ + q) * S_;
  const int t = threadIdx.x;
  __shared__ float redmax[8], redsum[8];

  f32x4 x = ((const f32x4*)rp)[t];
  float mx = fmaxf(fmaxf(x.x, x.y), fmaxf(x.z, x.w));
#pragma unroll
  for (int off = 16; off > 0; off >>= 1) mx = fmaxf(mx, __shfl_xor(mx, off, 32));
  if ((t & 31) == 0) redmax[t >> 5] = mx;
  __syncthreads();
  mx = redmax[0];
#pragma unroll
  for (int i = 1; i < 8; ++i) mx = fmaxf(mx, redmax[i]);

  float e0 = __expf(x.x - mx), e1 = __expf(x.y - mx);
  float e2 = __expf(x.z - mx), e3 = __expf(x.w - mx);
  float sum = (e0 + e1) + (e2 + e3);
#pragma unroll
  for (int off = 16; off > 0; off >>= 1) sum += __shfl_xor(sum, off, 32);
  if ((t & 31) == 0) redsum[t >> 5] = sum;
  __syncthreads();
  sum = redsum[0];
#pragma unroll
  for (int i = 1; i < 8; ++i) sum += redsum[i];

  const float inv = 1.0f / sum;
  f32x4 d = ((const f32x4*)dp)[t];
  f32x4 g = ((const f32x4*)gp)[t];
  f32x4 o;
  o.x = e0 * inv + d.x + g.x;  o.y = e1 * inv + d.y + g.y;
  o.z = e2 * inv + d.z + g.z;  o.w = e3 * inv + d.w + g.w;
  ((f32x4*)rp)[t] = o;
}

// ------------------------------------------------ xv = attn @ V  (per b,h)
// V tile staged in LDS *transposed* ([d][k], k contiguous, padded stride) so
// B-fragments are two ds_load_b128 per 16x32 fragment instead of 16 u16 loads.
#define VT_RS 264   // padded row stride in shorts (528 B -> 4-bank lane stride)
__global__ __launch_bounds__(128)
void attnv_kernel(const float* __restrict__ attn,
                  const unsigned short* __restrict__ Vb,
                  unsigned short* __restrict__ XVb) {
  const int bh = blockIdx.z, b = bh / H_, h = bh % H_;
  const int lane = threadIdx.x & 31, wave = threadIdx.x >> 5;
  const int lo = lane & 15, hi = lane >> 4;
  const int tileQ = (blockIdx.x * 4 + wave) * 64;
  const float* Abase = attn + (size_t)bh * S_ * S_;
  const unsigned short* Vbase = Vb + (size_t)b * S_ * E_ + h * HD_;

  __shared__ unsigned short Vt[HD_ * VT_RS];   // 64 x 264 shorts = 33 KB
  v8f acc[4][4] = {};
  for (int kb = 0; kb < S_; kb += 256) {
    __syncthreads();
    // cooperative transposed store: pair two k-rows -> packed b32 LDS stores
    unsigned int* base = (unsigned int*)Vt;
    for (int p = threadIdx.x; p < 1024; p += 128) {
      const int kp = p >> 3;          // k-pair index (k = 2*kp, 2*kp+1)
      const int c  = p & 7;           // column group: d = c*8 .. c*8+7
      const unsigned short* s0 = Vbase + (size_t)(kb + 2 * kp) * E_ + c * 8;
      u32x4 a = *(const u32x4*)s0;
      u32x4 bb2 = *(const u32x4*)(s0 + E_);
      const int d0 = c * 8;
      base[(d0 + 0) * (VT_RS / 2) + kp] = (a.x & 0xffffu) | (bb2.x << 16);
      base[(d0 + 1) * (VT_RS / 2) + kp] = (a.x >> 16) | (bb2.x & 0xffff0000u);
      base[(d0 + 2) * (VT_RS / 2) + kp] = (a.y & 0xffffu) | (bb2.y << 16);
      base[(d0 + 3) * (VT_RS / 2) + kp] = (a.y >> 16) | (bb2.y & 0xffff0000u);
      base[(d0 + 4) * (VT_RS / 2) + kp] = (a.z & 0xffffu) | (bb2.z << 16);
      base[(d0 + 5) * (VT_RS / 2) + kp] = (a.z >> 16) | (bb2.z & 0xffff0000u);
      base[(d0 + 6) * (VT_RS / 2) + kp] = (a.w & 0xffffu) | (bb2.w << 16);
      base[(d0 + 7) * (VT_RS / 2) + kp] = (a.w >> 16) | (bb2.w & 0xffff0000u);
    }
    __syncthreads();
#pragma unroll 1
    for (int k0 = 0; k0 < 256; k0 += 32) {
      Frag16 af[4], bf[4];
#pragma unroll
      for (int i = 0; i < 4; ++i) {
        const float* row = Abase + (size_t)(tileQ + i * 16 + lo) * S_ + kb + k0 + hi * 8;
        f32x4 f0 = ((const f32x4*)row)[0];
        f32x4 f1 = ((const f32x4*)row)[1];
        f32x4 f2 = ((const f32x4*)(row + 16))[0];
        f32x4 f3 = ((const f32x4*)(row + 16))[1];
        __builtin_prefetch(row + 256, 0, 1);   // next k-block of this attn row
        pack16(af[i], f0, f1, f2, f3);
      }
#pragma unroll
      for (int j = 0; j < 4; ++j) {
        const unsigned short* rowp = Vt + (size_t)(j * 16 + lo) * VT_RS + k0 + hi * 8;
        bf[j].q[0] = *(const u32x4*)rowp;        // k elems 8h..8h+7
        bf[j].q[1] = *(const u32x4*)(rowp + 16); // k elems 16+8h..16+8h+7
      }
#pragma unroll
      for (int i = 0; i < 4; ++i)
#pragma unroll
        for (int j = 0; j < 4; ++j)
          acc[i][j] = __builtin_amdgcn_wmma_f32_16x16x32_bf16(
              false, af[i].v, false, bf[j].v, (short)0, acc[i][j], false, false);
    }
  }
#pragma unroll
  for (int i = 0; i < 4; ++i)
#pragma unroll
    for (int j = 0; j < 4; ++j) {
      const int n = j * 16 + lo;
#pragma unroll
      for (int r = 0; r < 8; ++r) {
        const int qr = tileQ + i * 16 + hi * 8 + r;
        XVb[((size_t)b * S_ + qr) * E_ + h * HD_ + n] = f32_bf16(acc[i][j][r]);
      }
    }
}

extern "C" void kernel_launch(void* const* d_in, const int* in_sizes, int n_in,
                              void* d_out, int out_size, void* d_ws, size_t ws_size,
                              hipStream_t stream) {
  (void)in_sizes; (void)n_in; (void)out_size; (void)ws_size;
  const float* query = (const float*)d_in[0];
  const float* key   = (const float*)d_in[1];
  const float* value = (const float*)d_in[2];
  const float* mask  = (const float*)d_in[3];
  const float* gnn   = (const float*)d_in[4];
  const float* dist  = (const float*)d_in[5];
  const float* Wq = (const float*)d_in[6];  const float* bq = (const float*)d_in[7];
  const float* Wk = (const float*)d_in[8];  const float* bk = (const float*)d_in[9];
  const float* Wv = (const float*)d_in[10]; const float* bv = (const float*)d_in[11];
  const float* Wo = (const float*)d_in[12]; const float* bo = (const float*)d_in[13];

  float* out  = (float*)d_out;
  float* attn = out + (size_t)M_ * E_;     // second tuple output, also scratch

  char* ws = (char*)d_ws;
  const size_t MiB = 1024 * 1024;
  unsigned short* Wq_b = (unsigned short*)(ws + 0 * MiB);
  unsigned short* Wk_b = (unsigned short*)(ws + 2 * MiB);
  unsigned short* Wv_b = (unsigned short*)(ws + 4 * MiB);
  unsigned short* Wo_b = (unsigned short*)(ws + 6 * MiB);
  unsigned short* Qb   = (unsigned short*)(ws + 8 * MiB);
  unsigned short* Kb   = (unsigned short*)(ws + 24 * MiB);
  unsigned short* Vb   = (unsigned short*)(ws + 40 * MiB);
  unsigned short* XVb  = (unsigned short*)(ws + 56 * MiB);

  const int nP = E_ * E_ / 2;     // pairs per weight matrix
  dim3 cvtg((nP + 255) / 256), cvtb(256);
  cvt_f32_to_bf16_kernel<<<cvtg, cvtb, 0, stream>>>(Wq, (unsigned int*)Wq_b, nP);
  cvt_f32_to_bf16_kernel<<<cvtg, cvtb, 0, stream>>>(Wk, (unsigned int*)Wk_b, nP);
  cvt_f32_to_bf16_kernel<<<cvtg, cvtb, 0, stream>>>(Wv, (unsigned int*)Wv_b, nP);
  cvt_f32_to_bf16_kernel<<<cvtg, cvtb, 0, stream>>>(Wo, (unsigned int*)Wo_b, nP);

  dim3 gblk(128);
  dim3 ggrid(M_ / 128, E_ / 128);
  gemm_bf16_kernel<true,  false><<<ggrid, gblk, 0, stream>>>(query, Wq_b, bq, Qb, E_, E_);
  gemm_bf16_kernel<true,  false><<<ggrid, gblk, 0, stream>>>(key,   Wk_b, bk, Kb, E_, E_);
  gemm_bf16_kernel<true,  false><<<ggrid, gblk, 0, stream>>>(value, Wv_b, bv, Vb, E_, E_);

  scores_kernel<<<dim3(S_ / 128, S_ / 128, B_ * H_), gblk, 0, stream>>>(Qb, Kb, mask, attn);
  softmax_add_kernel<<<dim3(B_ * H_ * S_), dim3(256), 0, stream>>>(attn, dist, gnn);
  attnv_kernel<<<dim3(S_ / 256, 1, B_ * H_), gblk, 0, stream>>>(attn, Vb, XVb);

  gemm_bf16_kernel<false, true><<<ggrid, gblk, 0, stream>>>(XVb, Wo_b, bo, out, E_, E_);
}
